// Module_39556648796660
// MI455X (gfx1250) — matrix operand order repaired and checked by reading, not compiled, not run on hardware
//
#include <hip/hip_runtime.h>
#include <hip/hip_bf16.h>

// ---------------------------------------------------------------------------
// Spiking RNN with synaptic delays on MI455X (gfx1250, wave32, WMMA).
//
// Roofline: 500 sequential steps x (64x8192)*(8192x1024) f16 GEMM = 537 GFLOP.
// Static weights = 16 MB f16 -> L2 resident (192 MB). Per-step compute is ~us,
// so the whole scan runs in ONE persistent kernel; the delay structure gives a
// dependency distance of 2 steps, so a device-wide barrier every 2 steps.
// ---------------------------------------------------------------------------

typedef __attribute__((ext_vector_type(16))) _Float16 v16h;
typedef __attribute__((ext_vector_type(8)))  _Float16 v8h;
typedef __attribute__((ext_vector_type(8)))  float    v8f;

#define N_NEU   1024
#define NIN     256
#define NOUT    32
#define TSTEPS  500
#define BATCH   64
#define NDELAY  8
#define KDIM    (NDELAY * N_NEU)     // 8192
#define A_P     0.95f
#define A_MEM   0.9f
#define A_OUT   0.9f

// ---- workspace layout (bytes, all 256B-aligned) ----
static const size_t S_H1    = (size_t)TSTEPS * BATCH * N_NEU * 4;   // 131072000
static const size_t S_XH    = (size_t)BATCH * TSTEPS * NIN * 2;     // 16384000
static const size_t S_WINP  = (size_t)NIN * N_NEU * 2;              // 524288
static const size_t S_WOUTP = (size_t)N_NEU * NOUT * 2;             // 65536
static const size_t S_SWP   = (size_t)KDIM * N_NEU * 2;             // 16777216
static const size_t S_AREC  = (size_t)TSTEPS * BATCH * N_NEU * 2;   // 65536000
static const size_t S_SPK   = S_AREC;
static const size_t S_H2    = (size_t)TSTEPS * BATCH * NOUT * 4;    // 4096000
static const size_t S_MEM   = (size_t)BATCH * N_NEU * 4;
static const size_t S_WP    = S_MEM;

static const size_t OFF_H1    = 0;
static const size_t OFF_XH    = OFF_H1    + S_H1;
static const size_t OFF_WINP  = OFF_XH    + S_XH;
static const size_t OFF_WOUTP = OFF_WINP  + S_WINP;
static const size_t OFF_SWP   = OFF_WOUTP + S_WOUTP;
static const size_t OFF_AREC  = OFF_SWP   + S_SWP;
static const size_t OFF_SPK   = OFF_AREC  + S_AREC;
static const size_t OFF_H2    = OFF_SPK   + S_SPK;
static const size_t OFF_MEM   = OFF_H2    + S_H2;
static const size_t OFF_WP    = OFF_MEM   + S_MEM;
static const size_t OFF_BAR   = OFF_WP    + S_WP;

// ---------------------------------------------------------------------------
// f32 -> f16 elementwise convert (stage inputs for WMMA A operand)
// ---------------------------------------------------------------------------
__global__ void k_cvt_f16(const float* __restrict__ src,
                          _Float16* __restrict__ dst, int n) {
  int i = blockIdx.x * 256 + threadIdx.x;
  if (i < n) dst[i] = (_Float16)src[i];
}

// ---------------------------------------------------------------------------
// Pack a row-major f32 [K][N] matrix into WMMA B-fragment layout:
//   frag(nt, ks) lane L element j = B[ks*32 + 16*(L/16) + j][nt*16 + L%16]
// so each lane's 16 f16 (32 B) are contiguous -> two global_load_b128.
// ---------------------------------------------------------------------------
__global__ void k_pack_b(const float* __restrict__ src,
                         _Float16* __restrict__ dst, int K, int N) {
  int ksteps = K >> 5, ntiles = N >> 4;
  int id = blockIdx.x * 256 + threadIdx.x;
  int total = ntiles * ksteps * 32;
  if (id >= total) return;
  int lane = id & 31;
  int ks   = (id >> 5) % ksteps;
  int nt   = (id >> 5) / ksteps;
  int kbase = ks * 32 + 16 * (lane >> 4);
  int n     = nt * 16 + (lane & 15);
  _Float16* o = dst + (size_t)id * 16;
#pragma unroll
  for (int j = 0; j < 16; ++j)
    o[j] = (_Float16)src[(size_t)(kbase + j) * N + n];
}

// ---------------------------------------------------------------------------
// Build static_weights = dmap * sign[e] * |w[e][n]| as a packed B operand of
// shape [K=8192][N=1024] with K index = d*1024 + e.  16 MB f16 -> L2 resident.
// ---------------------------------------------------------------------------
__global__ void k_pack_sw(const float* __restrict__ w,
                          const float* __restrict__ sg,
                          const float* __restrict__ dmap,
                          _Float16* __restrict__ dst) {
  int id = blockIdx.x * 256 + threadIdx.x;         // 64 nt * 256 ks * 32 lanes
  if (id >= 64 * 256 * 32) return;
  int lane = id & 31;
  int ks   = (id >> 5) & 255;
  int nt   = id >> 13;
  int kbase = ks * 32 + 16 * (lane >> 4);
  int n     = nt * 16 + (lane & 15);
  _Float16* o = dst + (size_t)id * 16;
#pragma unroll
  for (int j = 0; j < 16; ++j) {
    int k = kbase + j, d = k >> 10, e = k & 1023;
    float v = dmap[((size_t)d << 20) + ((size_t)e << 10) + n] *
              sg[e] * fabsf(w[((size_t)e << 10) + n]);
    o[j] = (_Float16)v;
  }
}

// ---------------------------------------------------------------------------
// Generic f16 WMMA GEMM: A row-major [M][K] f16, Bp packed fragments,
// C f32 [Mout][N].  permMode 1 remaps row r=(b*T+t) -> out row (t*B+b) for h1.
// One 16x16 tile per wave, 4 waves per 128-thread workgroup.
// ---------------------------------------------------------------------------
__global__ void __launch_bounds__(128)
k_gemm(const _Float16* __restrict__ A, const _Float16* __restrict__ Bp,
       float* __restrict__ C, int M, int K, int N, int permMode) {
  int ksteps = K >> 5, ntiles = N >> 4;
  int wave = threadIdx.x >> 5, lane = threadIdx.x & 31;
  long tile = (long)blockIdx.x * 4 + wave;
  long ttot = (long)(M >> 4) * ntiles;
  if (tile >= ttot) return;                 // wave-uniform: EXEC stays all-1s
  int  nt = (int)(tile % ntiles);
  long mt = tile / ntiles;
  int  half = lane >> 4;
  const _Float16* arow = A + (size_t)(mt * 16 + (lane & 15)) * K + 16 * half;
  const _Float16* bptr = Bp + ((size_t)nt * ksteps * 32 + lane) * 16;
  v8f c = {};
  for (int ks = 0; ks < ksteps; ++ks) {
    union { v16h v; v8h h[2]; } ua, ub;
    ua.h[0] = *(const v8h*)(arow + (size_t)ks * 32);
    ua.h[1] = *(const v8h*)(arow + (size_t)ks * 32 + 8);
    ub.v    = *(const v16h*)(bptr + (size_t)ks * 512);
    c = __builtin_amdgcn_wmma_f32_16x16x32_f16(false, ua.v, false, ub.v,
                                               (short)0, c, false, false);
  }
  int ncol = nt * 16 + (lane & 15);
#pragma unroll
  for (int j = 0; j < 8; ++j) {
    long m = mt * 16 + j + 8 * half;
    long r = (permMode == 1) ? ((m % TSTEPS) * (long)BATCH + m / TSTEPS) : m;
    C[(size_t)r * N + ncol] = c[j];
  }
}

// ---------------------------------------------------------------------------
// Persistent recurrent scan. 256 WGs x 128 threads (4 waves).
// WG g owns output tile (mt = g&3 batch rows, nt = g>>2 neuron cols).
// Each wave covers K range [wave*2048, wave*2048+2048) = 2 delay buckets,
// 64 v_wmma_f32_16x16x32_f16 per wave per step; partials reduced via LDS;
// each wave then finalizes 2 of the 8 C rows and applies the membrane /
// plasticity update for its rows (each (b,n) has exactly one owner).
// Dependency distance is 2 steps (min delay = 1), so the device-wide atomic
// barrier only runs after odd steps: 250 barriers instead of 500.
// ---------------------------------------------------------------------------
__global__ void __launch_bounds__(128)
k_recurrent(const float* __restrict__ h1,        // [T][B][N] f32
            const _Float16* __restrict__ swp,    // packed [64][256][32][16]
            const float* __restrict__ p,         // [N]
            const int* __restrict__ delays,      // [8]
            _Float16* __restrict__ a_rec,        // [T][B][N] f16 (zeroed)
            _Float16* __restrict__ spk,          // [T][B][N] f16
            float* __restrict__ memv,            // [B][N] f32 (zeroed)
            float* __restrict__ wpv,             // [B][N] f32 (zeroed)
            unsigned* __restrict__ bar)          // [cnt, gen] (zeroed)
{
  __shared__ float red[4][32][9];            // pad 8->9: no 8-way bank clash
  const int wave = threadIdx.x >> 5, lane = threadIdx.x & 31;
  const int wg = blockIdx.x;
  const int mt = wg & 3, nt = wg >> 2;
  const int half = lane >> 4;
  const int bA = mt * 16 + (lane & 15);      // A row (batch) for this lane
  const int nC = nt * 16 + (lane & 15);      // C column (neuron) for this lane

  const int dly0 = delays[wave * 2 + 0];
  const int dly1 = delays[wave * 2 + 1];
  const float p_n  = p[nC];
  const float pdep = (p_n < 0.f) ? 1.f : 0.f;

  const _Float16* bbase =
      swp + (((size_t)nt * 256 + (size_t)wave * 64) * 32 + lane) * 16;

  for (int t = 0; t < TSTEPS; ++t) {
    int v0 = (t - 1 - dly0) % TSTEPS; if (v0 < 0) v0 += TSTEPS;
    int v1 = (t - 1 - dly1) % TSTEPS; if (v1 < 0) v1 += TSTEPS;

    v8f c = {};
    const _Float16* bptr = bbase;
#pragma unroll 1
    for (int dd = 0; dd < 2; ++dd) {
      const int tv = dd ? v1 : v0;
      const _Float16* abase =
          a_rec + ((size_t)tv * BATCH + bA) * N_NEU + 16 * half;
#pragma unroll 4
      for (int k2 = 0; k2 < 32; ++k2) {
        union { v16h v; v8h h[2]; } ua, ub;
        ua.h[0] = *(const v8h*)(abase + k2 * 32);
        ua.h[1] = *(const v8h*)(abase + k2 * 32 + 8);
        ub.v    = *(const v16h*)(bptr);
        bptr += 512;
        __builtin_prefetch(bptr, 0, 1);      // global_prefetch_b8 (L2-hot B)
        c = __builtin_amdgcn_wmma_f32_16x16x32_f16(false, ua.v, false, ub.v,
                                                   (short)0, c, false, false);
      }
    }

    // cross-wave K reduction through LDS
#pragma unroll
    for (int j = 0; j < 8; ++j) red[wave][lane][j] = c[j];
    __syncthreads();

    // each wave finalizes C rows {2w, 2w+1} (x two halves via lane>>4) and
    // applies the pointwise neuron update for those (b, nC) pairs
#pragma unroll
    for (int j2 = 0; j2 < 2; ++j2) {
      int j = wave * 2 + j2;
      float syn = red[0][lane][j] + red[1][lane][j] +
                  red[2][lane][j] + red[3][lane][j];
      int b = mt * 16 + j + 8 * half;
      size_t sidx = (size_t)b * N_NEU + nC;
      float m  = memv[sidx];
      float wp = wpv[sidx];
      float mthr = m - 1.f;
      float out  = (mthr > 0.f) ? 1.f : 0.f;      // spike (== rst)
      float wpn  = wp * A_P + out * p_n * (1.f + pdep * wp);
      size_t tidx = ((size_t)t * BATCH + b) * N_NEU + nC;
      spk[tidx]   = (_Float16)out;
      a_rec[tidx] = (_Float16)(out * (1.f + wpn));
      memv[sidx]  = A_MEM * m + h1[tidx] + syn - out;
      wpv[sidx]   = wpn;
    }

    if (t & 1) {
      // device-wide barrier every 2nd step (producer t .. consumer t+2)
      __threadfence();               // every storing thread flushes its own
      __syncthreads();               // ... before thread 0 signals arrival
      if (threadIdx.x == 0) {
        unsigned prev = __hip_atomic_fetch_add(&bar[0], 1u, __ATOMIC_ACQ_REL,
                                               __HIP_MEMORY_SCOPE_AGENT);
        if (prev == 255u) {
          __hip_atomic_store(&bar[0], 0u, __ATOMIC_RELAXED,
                             __HIP_MEMORY_SCOPE_AGENT);
          __hip_atomic_fetch_add(&bar[1], 1u, __ATOMIC_RELEASE,
                                 __HIP_MEMORY_SCOPE_AGENT);
        } else {
          while (__hip_atomic_load(&bar[1], __ATOMIC_ACQUIRE,
                                   __HIP_MEMORY_SCOPE_AGENT) <=
                 (unsigned)(t >> 1))
            __builtin_amdgcn_s_sleep(2);
        }
        __threadfence();
      }
      __syncthreads();
    } else {
      __syncthreads();               // protect LDS red reuse WG-locally
    }
  }
}

// ---------------------------------------------------------------------------
// Leaky output integrator: out[b][t][o] = A_OUT*out[b][t-1][o] + h2[t][b][o]
// ---------------------------------------------------------------------------
__global__ void k_outscan(const float* __restrict__ h2,
                          float* __restrict__ out) {
  int id = blockIdx.x * 256 + threadIdx.x;
  if (id >= BATCH * NOUT) return;
  int b = id >> 5, o = id & 31;
  float acc = 0.f;
  for (int t = 0; t < TSTEPS; ++t) {
    acc = A_OUT * acc + h2[((size_t)t * BATCH + b) * NOUT + o];
    out[((size_t)b * TSTEPS + t) * NOUT + o] = acc;
  }
}

// ---------------------------------------------------------------------------
extern "C" void kernel_launch(void* const* d_in, const int* in_sizes, int n_in,
                              void* d_out, int out_size, void* d_ws,
                              size_t ws_size, hipStream_t stream) {
  const float* inputs = (const float*)d_in[0];   // [B,T,NIN]
  const float* w      = (const float*)d_in[1];   // [N,N]
  const float* w_in   = (const float*)d_in[2];   // [NIN,N]
  const float* w_out  = (const float*)d_in[3];   // [N,NOUT]
  const float* w_sg   = (const float*)d_in[4];   // [N]
  const float* p      = (const float*)d_in[5];   // [N]
  const float* dmap   = (const float*)d_in[6];   // [D,N,N]
  const int*   delays = (const int*)d_in[7];     // [D]

  char* ws = (char*)d_ws;
  float*    h1    = (float*)(ws + OFF_H1);
  _Float16* xh    = (_Float16*)(ws + OFF_XH);
  _Float16* winp  = (_Float16*)(ws + OFF_WINP);
  _Float16* woutp = (_Float16*)(ws + OFF_WOUTP);
  _Float16* swp   = (_Float16*)(ws + OFF_SWP);
  _Float16* arec  = (_Float16*)(ws + OFF_AREC);
  _Float16* spk   = (_Float16*)(ws + OFF_SPK);
  float*    h2    = (float*)(ws + OFF_H2);
  float*    memv  = (float*)(ws + OFF_MEM);
  float*    wpv   = (float*)(ws + OFF_WP);
  unsigned* bar   = (unsigned*)(ws + OFF_BAR);

  // Per-call state init (graph-capture safe; keeps replays deterministic).
  hipMemsetAsync(arec, 0, S_AREC, stream);   // wrap-around reads must be 0
  hipMemsetAsync(memv, 0, S_MEM, stream);
  hipMemsetAsync(wpv,  0, S_WP,  stream);
  hipMemsetAsync(bar,  0, 256,   stream);

  // Stage operands.
  k_cvt_f16<<<(BATCH * TSTEPS * NIN + 255) / 256, 256, 0, stream>>>(
      inputs, xh, BATCH * TSTEPS * NIN);
  k_pack_b<<<((N_NEU / 16) * (NIN / 32) * 32 + 255) / 256, 256, 0, stream>>>(
      w_in, winp, NIN, N_NEU);
  k_pack_b<<<((NOUT / 16) * (N_NEU / 32) * 32 + 255) / 256, 256, 0, stream>>>(
      w_out, woutp, N_NEU, NOUT);
  k_pack_sw<<<(64 * 256 * 32) / 256, 256, 0, stream>>>(w, w_sg, dmap, swp);

  // h1 = inputs @ w_in : M=32000, K=256, N=1024 (rows permuted b*T+t -> t*B+b)
  k_gemm<<<((32000 / 16) * (N_NEU / 16) + 3) / 4, 128, 0, stream>>>(
      xh, winp, h1, 32000, NIN, N_NEU, 1);

  // 500-step recurrent scan (persistent, grid barrier every 2 steps).
  k_recurrent<<<256, 128, 0, stream>>>(h1, swp, p, delays, arec, spk,
                                       memv, wpv, bar);

  // h2 = spk @ w_out : M=32000, K=1024, N=32
  k_gemm<<<((32000 / 16) * (NOUT / 16) + 3) / 4, 128, 0, stream>>>(
      spk, woutp, h2, 32000, N_NEU, NOUT, 0);

  k_outscan<<<8, 256, 0, stream>>>(h2, (float*)d_out);
}
